// HierarchicalJointModel_80075370266984
// MI455X (gfx1250) — matrix operand order repaired and checked by reading, not compiled, not run on hardware
//
#include <hip/hip_runtime.h>
#include <hip/hip_bf16.h>

// ---------------------------------------------------------------------------
// CDNA5 (gfx1250) hierarchical GRU implementation.
// All matrix work (input projections + recurrent matmuls of all three GRU
// levels) goes through v_wmma_f32_16x16x32_f16. h is kept in LDS (f16,
// ping-pong buffered) per 16-row batch tile; one barrier per time step.
// Fragment loads are explicit 16-byte vector loads (no per-lane predication)
// so they lower to global_load_b128 / ds_load_b128.
// ---------------------------------------------------------------------------

typedef __attribute__((ext_vector_type(16))) _Float16 v16h;
typedef __attribute__((ext_vector_type(8)))  _Float16 v8h;
typedef __attribute__((ext_vector_type(8)))  float    v8f;

#define HH 256   // hidden size

__device__ __forceinline__ float sigmoidf_(float x) { return 1.f / (1.f + __expf(-x)); }

// A fragment: 16x32 f16, row-major source, lane holds row m=lane&15.
// K layout per ISA: lane-half hi holds K = hi*8+{0..7} and 16+hi*8+{0..7}.
// p points at &row[k0]; both chunks are 16B aligned.
__device__ __forceinline__ v16h load_a_frag(const _Float16* p, int hi) {
  v8h lo = *(const v8h*)(p + hi * 8);        // K = hi*8 + 0..7
  v8h hs = *(const v8h*)(p + 16 + hi * 8);   // K = 16 + hi*8 + 0..7
  return __builtin_shufflevector(lo, hs, 0, 1, 2, 3, 4, 5, 6, 7,
                                 8, 9, 10, 11, 12, 13, 14, 15);
}

// B fragment: 32x16 f16 where B = W^T (W row-major [N x K]).
// Lane holds col n=lane&15; K = hi*8+{0..7} and 16+hi*8+{0..7} along W's K dim.
// p = &W[n_global*ldK + k0].
__device__ __forceinline__ v16h load_b_frag(const _Float16* p, int hi) {
  const _Float16* q = p + hi * 8;
  v8h lo = *(const v8h*)q;
  v8h hs = *(const v8h*)(q + 16);
  return __builtin_shufflevector(lo, hs, 0, 1, 2, 3, 4, 5, 6, 7,
                                 8, 9, 10, 11, 12, 13, 14, 15);
}

__device__ __forceinline__ v8f wmma16(v16h a, v16h b, v8f c) {
  return __builtin_amdgcn_wmma_f32_16x16x32_f16(false, a, false, b, (short)0, c,
                                                false, false);
}

// ---------------------------------------------------------------------------
// Generic bidirectional GRU scan over a 16-row batch tile.
// grid = (rowTiles, 2 dirs), block = 256 (8 wave32).
// xin: [rowsPad, T, din] f16 with rowsPad a multiple of 16 (pad rows readable).
// w*: f16 [3H x din] / [3H x H].  hout: [rows, 2H] f16.
// ---------------------------------------------------------------------------
__global__ void __launch_bounds__(256) gru_scan_kernel(
    const _Float16* __restrict__ xin,
    const _Float16* __restrict__ wih_f, const _Float16* __restrict__ wih_b,
    const _Float16* __restrict__ whh_f, const _Float16* __restrict__ whh_b,
    const float* __restrict__ bih_f, const float* __restrict__ bih_b,
    const float* __restrict__ bhh_f, const float* __restrict__ bhh_b,
    const int* __restrict__ lengths, _Float16* __restrict__ hout,
    int rows, int T, int din) {
  const int dir = blockIdx.y;
  const bool reverse = (dir == 1);
  const _Float16* wih = dir ? wih_b : wih_f;
  const _Float16* whh = dir ? whh_b : whh_f;
  const float* bih = dir ? bih_b : bih_f;
  const float* bhh = dir ? bhh_b : bhh_f;

  const int tile = blockIdx.x;
  const int tid  = threadIdx.x;
  const int wid  = tid >> 5;
  const int lane = tid & 31;
  const int ml   = lane & 15;   // A row / B col within tile
  const int hi   = lane >> 4;

  __shared__ _Float16 hbuf[2][16 * HH];
  __shared__ int lsm[16];

  for (int idx = tid; idx < 16 * HH; idx += 256) {
    hbuf[0][idx] = (_Float16)0.f;
    hbuf[1][idx] = (_Float16)0.f;
  }
  if (tid < 16) {
    int rg = tile * 16 + tid;
    lsm[tid] = (rg < rows) ? (lengths ? lengths[rg] : T) : 0;
  }
  __syncthreads();

  // Per-lane gate biases for this lane's two column chunks (r,z,n).
  float bi[2][3], bh[2][3];
#pragma unroll
  for (int cc = 0; cc < 2; ++cc) {
    int col = (wid + cc * 8) * 16 + ml;
#pragma unroll
    for (int g = 0; g < 3; ++g) {
      bi[cc][g] = bih[g * HH + col];
      bh[cc][g] = bhh[g * HH + col];
    }
  }

  const int rg = tile * 16 + ml;   // this lane's A row (pad rows are readable)

  int cur = 0;
  for (int s = 0; s < T; ++s) {
    const int t = reverse ? (T - 1 - s) : s;
    const _Float16* xrow = xin + ((size_t)rg * T + t) * din;
    const _Float16* hcur = hbuf[cur];
    _Float16*       hnxt = hbuf[cur ^ 1];

    for (int cc = 0; cc < 2; ++cc) {
      const int j    = wid + cc * 8;   // column tile 0..15 of H
      const int coln = j * 16 + ml;    // this lane's B column (global within H)

      v8f gi0 = {}, gi1 = {}, gi2 = {};
      v8f gh0 = {}, gh1 = {}, gh2 = {};

      // gi = x_t @ Wih^T (three gate column blocks)
      for (int k0 = 0; k0 < din; k0 += 32) {
        v16h a = load_a_frag(xrow + k0, hi);
        gi0 = wmma16(a, load_b_frag(wih + ((size_t)(0 * HH + coln)) * din + k0, hi), gi0);
        gi1 = wmma16(a, load_b_frag(wih + ((size_t)(1 * HH + coln)) * din + k0, hi), gi1);
        gi2 = wmma16(a, load_b_frag(wih + ((size_t)(2 * HH + coln)) * din + k0, hi), gi2);
      }
      // gh = h @ Whh^T (A sourced from LDS)
      for (int k0 = 0; k0 < HH; k0 += 32) {
        v16h a = load_a_frag(hcur + ml * HH + k0, hi);
        gh0 = wmma16(a, load_b_frag(whh + ((size_t)(0 * HH + coln)) * HH + k0, hi), gh0);
        gh1 = wmma16(a, load_b_frag(whh + ((size_t)(1 * HH + coln)) * HH + k0, hi), gh1);
        gh2 = wmma16(a, load_b_frag(whh + ((size_t)(2 * HH + coln)) * HH + k0, hi), gh2);
      }

      // Gate epilogue (PyTorch order r,z,n). C layout: VGPR d -> row d+8*hi.
#pragma unroll
      for (int d = 0; d < 8; ++d) {
        int row = d + hi * 8;
        float rr = sigmoidf_(gi0[d] + gh0[d] + bi[cc][0] + bh[cc][0]);
        float zz = sigmoidf_(gi1[d] + gh1[d] + bi[cc][1] + bh[cc][1]);
        float nn = tanhf(gi2[d] + bi[cc][2] + rr * (gh2[d] + bh[cc][2]));
        float hold = (float)hcur[row * HH + coln];
        float hnew = (1.f - zz) * nn + zz * hold;
        bool  msk  = t < lsm[row];
        hnxt[row * HH + coln] = (_Float16)(msk ? hnew : hold);
      }
    }
    __syncthreads();
    cur ^= 1;
  }

  const _Float16* hf = hbuf[cur];
  for (int idx = tid; idx < 16 * HH; idx += 256) {
    int row = idx >> 8;
    int col = idx & 255;
    int rgw = tile * 16 + row;
    if (rgw < rows) hout[(size_t)rgw * (2 * HH) + dir * HH + col] = hf[idx];
  }
}

// ---------------------------------------------------------------------------
// Support kernels
// ---------------------------------------------------------------------------
__global__ void cvt_pad_kernel(const float* __restrict__ src, _Float16* __restrict__ dst,
                               int N, int K, int Kp) {
  int idx = blockIdx.x * blockDim.x + threadIdx.x;
  if (idx < N * Kp) {
    int r = idx / Kp, c = idx % Kp;
    dst[idx] = (c < K) ? (_Float16)src[(size_t)r * K + c] : (_Float16)0.f;
  }
}

__global__ void gather_embed_kernel(const int* __restrict__ tok,
                                    const float* __restrict__ emb,
                                    _Float16* __restrict__ xin) {
  int b = blockIdx.x;               // flat (sentence, t)
  int id = tok[b];
  xin[(size_t)b * 256 + threadIdx.x] = (_Float16)emb[(size_t)id * 256 + threadIdx.x];
}

__global__ void sent_len_kernel(const int* __restrict__ tok, int* __restrict__ s_len,
                                int nsent, int T) {
  int i = blockIdx.x * blockDim.x + threadIdx.x;
  if (i < nsent) {
    int c = 0;
    for (int t = 0; t < T; ++t) c += (tok[(size_t)i * T + t] != 0);
    s_len[i] = c;
  }
}

__global__ void rev_len_kernel(const int* __restrict__ s_len, int* __restrict__ r_len,
                               int npr, int S) {
  int i = blockIdx.x * blockDim.x + threadIdx.x;
  if (i < npr) {
    int c = 0;
    for (int s = 0; s < S; ++s) c += (s_len[i * S + s] > 0);
    r_len[i] = c;
  }
}

// Build product-GRU input [16 (pad of 8), 16, 544]:
// [rev_h (512) | normalized user feats (20) | pad]. Pad rows are zero-filled.
__global__ void build_pin_kernel(const _Float16* __restrict__ rev_h,
                                 const float* __restrict__ user_feats,
                                 const float* __restrict__ uw,
                                 _Float16* __restrict__ p_in, int npr) {
  int pr = blockIdx.x;              // 0..255 (rows 128..255 are pad)
  bool real = pr < npr;
  int p = pr >> 4, r = pr & 15;
  for (int c = threadIdx.x; c < 544; c += blockDim.x) {
    _Float16 v = (_Float16)0.f;
    if (real) {
      if (c < 512) {
        v = rev_h[(size_t)pr * 512 + c];
      } else if (c < 532) {
        int u = c - 512;
        float ss = 0.f;
        for (int rr = 0; rr < 16; ++rr) {
          float x = user_feats[(size_t)(p * 16 + rr) * 20 + u];
          ss += x * x;
        }
        float nrm = fmaxf(sqrtf(ss), 1e-12f);
        v = (_Float16)(user_feats[(size_t)pr * 20 + u] / nrm * uw[u]);
      }
    }
    p_in[(size_t)pr * 544 + c] = v;
  }
}

// Review head: r_stars[b] = elu(x @ W1^T + b1) @ W2^T + b2 ; one block per (p,r)
__global__ void rff_kernel(const _Float16* __restrict__ p_in,
                           const float* __restrict__ w1, const float* __restrict__ b1,
                           const float* __restrict__ w2, const float* __restrict__ b2,
                           float* __restrict__ out) {
  int b = blockIdx.x;
  int i = threadIdx.x;                 // 0..127
  const _Float16* x = p_in + (size_t)b * 544;
  float acc = b1[i];
  for (int k = 0; k < 532; ++k) acc += w1[(size_t)i * 532 + k] * (float)x[k];
  float h1 = acc > 0.f ? acc : expm1f(acc);
  __shared__ float red[128];
  red[i] = h1 * w2[i];
  __syncthreads();
  for (int sft = 64; sft > 0; sft >>= 1) {
    if (i < sft) red[i] += red[i + sft];
    __syncthreads();
  }
  if (i == 0) out[72 + b] = red[0] + b2[0];
}

// Product head: p_stars[b, 0..8]
__global__ void pff_kernel(const _Float16* __restrict__ hprod,
                           const float* __restrict__ w1, const float* __restrict__ b1,
                           const float* __restrict__ w2, const float* __restrict__ b2,
                           float* __restrict__ out) {
  int b = blockIdx.x;
  int i = threadIdx.x;                 // 0..127
  const _Float16* x = hprod + (size_t)b * 512;
  float acc = b1[i];
  for (int k = 0; k < 512; ++k) acc += w1[(size_t)i * 512 + k] * (float)x[k];
  float h1 = acc > 0.f ? acc : expm1f(acc);
  __shared__ float hs[128];
  hs[i] = h1;
  __syncthreads();
  if (i < 9) {
    float o = b2[i];
    for (int k = 0; k < 128; ++k) o += w2[i * 128 + k] * hs[k];
    out[b * 9 + i] = o;
  }
}

// ---------------------------------------------------------------------------
extern "C" void kernel_launch(void* const* d_in, const int* in_sizes, int n_in,
                              void* d_out, int out_size, void* d_ws, size_t ws_size,
                              hipStream_t stream) {
  (void)in_sizes; (void)n_in; (void)out_size; (void)ws_size;
  const int P = 8, R = 16, S = 12, T = 32, E = 256;
  const int NS = P * R * S;   // 1536 sentences
  const int NPR = P * R;      // 128 reviews

  const int*   tok        = (const int*)d_in[0];
  const float* user_feats = (const float*)d_in[1];
  const float* emb        = (const float*)d_in[2];
  const float* uw         = (const float*)d_in[3];
  // GRU params: per level (s,r,p), per dir (f,b): wih, whh, bih, bhh
  const float* s_wih_f = (const float*)d_in[4];
  const float* s_whh_f = (const float*)d_in[5];
  const float* s_bih_f = (const float*)d_in[6];
  const float* s_bhh_f = (const float*)d_in[7];
  const float* s_wih_b = (const float*)d_in[8];
  const float* s_whh_b = (const float*)d_in[9];
  const float* s_bih_b = (const float*)d_in[10];
  const float* s_bhh_b = (const float*)d_in[11];
  const float* r_wih_f = (const float*)d_in[12];
  const float* r_whh_f = (const float*)d_in[13];
  const float* r_bih_f = (const float*)d_in[14];
  const float* r_bhh_f = (const float*)d_in[15];
  const float* r_wih_b = (const float*)d_in[16];
  const float* r_whh_b = (const float*)d_in[17];
  const float* r_bih_b = (const float*)d_in[18];
  const float* r_bhh_b = (const float*)d_in[19];
  const float* p_wih_f = (const float*)d_in[20];
  const float* p_whh_f = (const float*)d_in[21];
  const float* p_bih_f = (const float*)d_in[22];
  const float* p_bhh_f = (const float*)d_in[23];
  const float* p_wih_b = (const float*)d_in[24];
  const float* p_whh_b = (const float*)d_in[25];
  const float* p_bih_b = (const float*)d_in[26];
  const float* p_bhh_b = (const float*)d_in[27];
  const float* rff_w1 = (const float*)d_in[28];
  const float* rff_b1 = (const float*)d_in[29];
  const float* rff_w2 = (const float*)d_in[30];
  const float* rff_b2 = (const float*)d_in[31];
  const float* pff_w1 = (const float*)d_in[32];
  const float* pff_b1 = (const float*)d_in[33];
  const float* pff_w2 = (const float*)d_in[34];
  const float* pff_b2 = (const float*)d_in[35];

  float* out = (float*)d_out;

  // Workspace bump allocator (256B aligned).
  char* ws = (char*)d_ws;
  size_t off = 0;
  auto alloc = [&](size_t bytes) -> void* {
    void* p = ws + off;
    off += (bytes + 255) & ~(size_t)255;
    return p;
  };

  _Float16* xin_s  = (_Float16*)alloc((size_t)NS * T * E * 2);    // 25.2 MB
  _Float16* sent_h = (_Float16*)alloc((size_t)NS * 512 * 2);      // 1.6 MB
  _Float16* rev_h  = (_Float16*)alloc((size_t)NPR * 512 * 2);
  _Float16* p_in   = (_Float16*)alloc((size_t)256 * 544 * 2);     // 16 rows x 16 t (pad)
  _Float16* prod_h = (_Float16*)alloc((size_t)16 * 512 * 2);
  int* s_len = (int*)alloc((size_t)NS * 4);
  int* r_len = (int*)alloc((size_t)NPR * 4);

  _Float16* sWf = (_Float16*)alloc((size_t)768 * 256 * 2);
  _Float16* sWb = (_Float16*)alloc((size_t)768 * 256 * 2);
  _Float16* sUf = (_Float16*)alloc((size_t)768 * 256 * 2);
  _Float16* sUb = (_Float16*)alloc((size_t)768 * 256 * 2);
  _Float16* rWf = (_Float16*)alloc((size_t)768 * 512 * 2);
  _Float16* rWb = (_Float16*)alloc((size_t)768 * 512 * 2);
  _Float16* rUf = (_Float16*)alloc((size_t)768 * 256 * 2);
  _Float16* rUb = (_Float16*)alloc((size_t)768 * 256 * 2);
  _Float16* pWf = (_Float16*)alloc((size_t)768 * 544 * 2);
  _Float16* pWb = (_Float16*)alloc((size_t)768 * 544 * 2);
  _Float16* pUf = (_Float16*)alloc((size_t)768 * 256 * 2);
  _Float16* pUb = (_Float16*)alloc((size_t)768 * 256 * 2);

  auto cvt = [&](const float* src, _Float16* dst, int N, int K, int Kp) {
    int total = N * Kp;
    cvt_pad_kernel<<<dim3((total + 255) / 256), dim3(256), 0, stream>>>(src, dst, N, K, Kp);
  };
  cvt(s_wih_f, sWf, 768, 256, 256);  cvt(s_wih_b, sWb, 768, 256, 256);
  cvt(s_whh_f, sUf, 768, 256, 256);  cvt(s_whh_b, sUb, 768, 256, 256);
  cvt(r_wih_f, rWf, 768, 512, 512);  cvt(r_wih_b, rWb, 768, 512, 512);
  cvt(r_whh_f, rUf, 768, 256, 256);  cvt(r_whh_b, rUb, 768, 256, 256);
  cvt(p_wih_f, pWf, 768, 532, 544);  cvt(p_wih_b, pWb, 768, 532, 544);
  cvt(p_whh_f, pUf, 768, 256, 256);  cvt(p_whh_b, pUb, 768, 256, 256);

  sent_len_kernel<<<dim3((NS + 255) / 256), dim3(256), 0, stream>>>(tok, s_len, NS, T);
  rev_len_kernel<<<dim3(1), dim3(128), 0, stream>>>(s_len, r_len, NPR, S);
  gather_embed_kernel<<<dim3(NS * T), dim3(256), 0, stream>>>(tok, emb, xin_s);

  // Sentence-level bidirectional GRU: 96 row tiles x 2 dirs
  gru_scan_kernel<<<dim3(NS / 16, 2), dim3(256), 0, stream>>>(
      xin_s, sWf, sWb, sUf, sUb, s_bih_f, s_bih_b, s_bhh_f, s_bhh_b,
      s_len, sent_h, NS, T, E);

  // Review-level: input [128, 12, 512]
  gru_scan_kernel<<<dim3(NPR / 16, 2), dim3(256), 0, stream>>>(
      sent_h, rWf, rWb, rUf, rUb, r_bih_f, r_bih_b, r_bhh_f, r_bhh_b,
      r_len, rev_h, NPR, S, 512);

  build_pin_kernel<<<dim3(256), dim3(256), 0, stream>>>(rev_h, user_feats, uw, p_in, NPR);

  // Product-level: input [16 (8 real), 16, 544], full length (no packing)
  gru_scan_kernel<<<dim3(1, 2), dim3(256), 0, stream>>>(
      p_in, pWf, pWb, pUf, pUb, p_bih_f, p_bih_b, p_bhh_f, p_bhh_b,
      (const int*)nullptr, prod_h, P, R, 544);

  rff_kernel<<<dim3(NPR), dim3(128), 0, stream>>>(p_in, rff_w1, rff_b1, rff_w2, rff_b2, out);
  pff_kernel<<<dim3(P), dim3(128), 0, stream>>>(prod_h, pff_w1, pff_b1, pff_w2, pff_b2, out);
}
